// DeltaNetModel_3624952398808
// MI455X (gfx1250) — matrix operand order repaired and checked by reading, not compiled, not run on hardware
//
#include <hip/hip_runtime.h>
#include <hip/hip_bf16.h>

// ---------------------------------------------------------------------------
// Types for CDNA5 WMMA (gfx1250, wave32)
// ---------------------------------------------------------------------------
typedef __attribute__((ext_vector_type(16))) __bf16        v16bf;
typedef __attribute__((ext_vector_type(8)))  float         v8f;
typedef __attribute__((ext_vector_type(8)))  unsigned short ushort8;
typedef __attribute__((ext_vector_type(16))) unsigned short ushort16;

static __device__ __forceinline__ unsigned short f32_to_bf16(float f) {
    unsigned int u = __builtin_bit_cast(unsigned int, f);
    unsigned int lsb = (u >> 16) & 1u;
    u += 0x7fffu + lsb;               // round-to-nearest-even
    return (unsigned short)(u >> 16);
}

static __device__ __forceinline__ float silu(float x) {
    return x / (1.f + __expf(-x));
}

// ---------------------------------------------------------------------------
// CDNA5 async global->LDS DMA (ASYNCcnt-tracked), inline asm per ISA 15.18.3.
// LDS operand: low 32 bits of the generic pointer = LDS byte offset
// (the LDS flat aperture base has zero low 32 bits).
// ---------------------------------------------------------------------------
static __device__ __forceinline__ unsigned lds_off(const void* p) {
    return (unsigned)(size_t)p;
}
static __device__ __forceinline__ void async_load_b128(unsigned ldsOff, const void* g) {
    asm volatile("global_load_async_to_lds_b128 %0, %1, off"
                 :: "v"(ldsOff), "v"((unsigned long long)(size_t)g)
                 : "memory");
}
static __device__ __forceinline__ void wait_asynccnt_4() {
    asm volatile("s_wait_asynccnt 4" ::: "memory");
}
static __device__ __forceinline__ void wait_asynccnt_0() {
    asm volatile("s_wait_asynccnt 0" ::: "memory");
}

// ---------------------------------------------------------------------------
// Weight convert + transpose: in f32 [K,N] -> out bf16 [N,K]
// ---------------------------------------------------------------------------
__global__ void transpose_to_bf16_kernel(const float* __restrict__ in,
                                         unsigned short* __restrict__ out,
                                         int K, int N) {
    __shared__ float tile[32][33];
    const int kb = blockIdx.y * 32;
    const int nb = blockIdx.x * 32;
    const int tx = threadIdx.x & 31;
    const int ty = threadIdx.x >> 5;      // 0..7
    #pragma unroll
    for (int i = 0; i < 32; i += 8) {
        int k = kb + ty + i, n = nb + tx;
        tile[ty + i][tx] = (k < K && n < N) ? in[(size_t)k * N + n] : 0.f;
    }
    __syncthreads();
    #pragma unroll
    for (int i = 0; i < 32; i += 8) {
        int n = nb + ty + i, k = kb + tx;
        if (n < N && k < K)
            out[(size_t)n * K + k] = f32_to_bf16(tile[tx][ty + i]);
    }
}

// ---------------------------------------------------------------------------
// Embedding gather: x = emb[tokens], f32 + bf16 copies
// ---------------------------------------------------------------------------
__global__ void embed_kernel(const int* __restrict__ tokens,
                             const float* __restrict__ emb,
                             float* __restrict__ xf,
                             unsigned short* __restrict__ xb, int D) {
    const int m = blockIdx.x;
    const int tok = tokens[m];
    const float* src = emb + (size_t)tok * D;
    for (int d = threadIdx.x; d < D; d += blockDim.x) {
        float v = src[d];
        xf[(size_t)m * D + d] = v;
        xb[(size_t)m * D + d] = f32_to_bf16(v);
    }
}

// ---------------------------------------------------------------------------
// WMMA GEMM: C[M,N](f32) = act(A[M,K](bf16) @ Bt[N,K](bf16)^T)
// Block tile 128x128, K-step 32, 8 waves, each wave 64x32 (4x2 WMMA tiles).
// Double-buffered LDS, tiles staged with GLOBAL_LOAD_ASYNC_TO_LDS_B128:
// per thread 4 async b128 ops per tile; next tile's DMA overlaps WMMA.
// ACT: 0=none, 1=silu.  WB: also write bf16 copy of C.
// All dims are multiples of the tile sizes for this model.
// ---------------------------------------------------------------------------
template <int ACT, bool WB>
__global__ __launch_bounds__(256) void gemm_bf16_kernel(
    const unsigned short* __restrict__ A,    // [M,K]
    const unsigned short* __restrict__ Bt,   // [N,K]
    float* __restrict__ C,                   // [M,N]
    unsigned short* __restrict__ Cbf,        // [M,N] (if WB)
    int M, int N, int K) {
    __shared__ unsigned short Asm[2][128][40];  // [buf][m][k], 80B row stride
    __shared__ unsigned short Bsm[2][128][40];  // [buf][n][k]

    const int tid  = threadIdx.x;
    const int lane = tid & 31;
    const int wave = tid >> 5;               // 0..7
    const int bm = blockIdx.y * 128;
    const int bn = blockIdx.x * 128;
    const int wm = (wave & 1) * 64;          // wave row base within block
    const int wn = (wave >> 1) * 32;         // wave col base within block
    const int lrow = lane & 15;
    const int hi16 = lane >> 4;              // 0 for lanes 0-15, 1 for 16-31

    // this thread's two (row, col) staging slots (same for A and B tiles)
    const int row0 = tid >> 2;               // chunk = tid
    const int col0 = (tid & 3) * 8;
    const int row1 = (tid + 256) >> 2;       // chunk = tid + 256
    const int col1 = ((tid + 256) & 3) * 8;

    v8f acc[4][2] = {};

    const int nk = K / 32;

    // prologue: stage tile 0 into buffer 0 (4 async b128 per thread)
    {
        async_load_b128(lds_off(&Asm[0][row0][col0]), &A[(size_t)(bm + row0) * K + col0]);
        async_load_b128(lds_off(&Asm[0][row1][col1]), &A[(size_t)(bm + row1) * K + col1]);
        async_load_b128(lds_off(&Bsm[0][row0][col0]), &Bt[(size_t)(bn + row0) * K + col0]);
        async_load_b128(lds_off(&Bsm[0][row1][col1]), &Bt[(size_t)(bn + row1) * K + col1]);
    }

    for (int it = 0; it < nk; ++it) {
        const int buf = it & 1;
        const int k0n = (it + 1) * 32;
        if (it + 1 < nk) {
            // stage next tile into the other buffer (overlaps with compute below)
            const int nb_ = buf ^ 1;
            async_load_b128(lds_off(&Asm[nb_][row0][col0]), &A[(size_t)(bm + row0) * K + k0n + col0]);
            async_load_b128(lds_off(&Asm[nb_][row1][col1]), &A[(size_t)(bm + row1) * K + k0n + col1]);
            async_load_b128(lds_off(&Bsm[nb_][row0][col0]), &Bt[(size_t)(bn + row0) * K + k0n + col0]);
            async_load_b128(lds_off(&Bsm[nb_][row1][col1]), &Bt[(size_t)(bn + row1) * K + k0n + col1]);
            wait_asynccnt_4();   // current tile's 4 ops complete (in-order)
        } else {
            wait_asynccnt_0();
        }
        __syncthreads();         // all waves' DMA for this tile visible

        // ---- A fragments per ISA layout (16-bit A 16x32) ----
        // lanes 0-15 : M=lane,    elems = K[0..7],  K[16..23]
        // lanes 16-31: M=lane-16, elems = K[8..15], K[24..31]
        v16bf afrag[4];
        #pragma unroll
        for (int mt = 0; mt < 4; ++mt) {
            const unsigned short* base = &Asm[buf][wm + mt * 16 + lrow][hi16 * 8];
            ushort8 lo = *(const ushort8*)(base);
            ushort8 hi = *(const ushort8*)(base + 16);
            ushort16 cat = __builtin_shufflevector(lo, hi, 0, 1, 2, 3, 4, 5, 6, 7,
                                                   8, 9, 10, 11, 12, 13, 14, 15);
            afrag[mt] = __builtin_bit_cast(v16bf, cat);
        }
        // ---- B fragments (16-bit B 32x16) ----
        // lanes 0-15 : N=lane,    elems = K[0..7],  K[16..23]
        // lanes 16-31: N=lane-16, elems = K[8..15], K[24..31]
        v16bf bfrag[2];
        #pragma unroll
        for (int nt = 0; nt < 2; ++nt) {
            const unsigned short* base = &Bsm[buf][wn + nt * 16 + lrow][hi16 * 8];
            ushort8 lo = *(const ushort8*)(base);
            ushort8 hi = *(const ushort8*)(base + 16);
            ushort16 cat = __builtin_shufflevector(lo, hi, 0, 1, 2, 3, 4, 5, 6, 7,
                                                   8, 9, 10, 11, 12, 13, 14, 15);
            bfrag[nt] = __builtin_bit_cast(v16bf, cat);
        }
        // ---- 8 WMMAs per K-step ----
        #pragma unroll
        for (int mt = 0; mt < 4; ++mt)
            #pragma unroll
            for (int nt = 0; nt < 2; ++nt)
                acc[mt][nt] = __builtin_amdgcn_wmma_f32_16x16x32_bf16(
                    false, afrag[mt], false, bfrag[nt],
                    (short)0, acc[mt][nt], false, false);
        __syncthreads();         // done reading buf before its next overwrite
    }

    // ---- epilogue: C/D layout: VGPR r -> M=r (lanes 0-15), M=r+8 (16-31) ----
    #pragma unroll
    for (int mt = 0; mt < 4; ++mt) {
        #pragma unroll
        for (int nt = 0; nt < 2; ++nt) {
            #pragma unroll
            for (int r = 0; r < 8; ++r) {
                int row = bm + wm + mt * 16 + r + hi16 * 8;
                int col = bn + wn + nt * 16 + lrow;
                float v = acc[mt][nt][r];
                if (ACT == 1) v = silu(v);
                C[(size_t)row * N + col] = v;
                if (WB) Cbf[(size_t)row * N + col] = f32_to_bf16(v);
            }
        }
    }
}

// ---------------------------------------------------------------------------
// beta = sigmoid(x @ Wb) : one block per row
// ---------------------------------------------------------------------------
__global__ void beta_kernel(const float* __restrict__ x,
                            const float* __restrict__ Wb,
                            float* __restrict__ beta, int D) {
    __shared__ float red[256];
    const size_t row = (size_t)blockIdx.x * D;
    float s = 0.f;
    for (int d = threadIdx.x; d < D; d += 256) s += x[row + d] * Wb[d];
    red[threadIdx.x] = s;
    __syncthreads();
    for (int off = 128; off > 0; off >>= 1) {
        if (threadIdx.x < off) red[threadIdx.x] += red[threadIdx.x + off];
        __syncthreads();
    }
    if (threadIdx.x == 0) beta[blockIdx.x] = 1.f / (1.f + __expf(-red[0]));
}

// ---------------------------------------------------------------------------
// In-place row L2 norm (eps 1e-6): one block per row
// ---------------------------------------------------------------------------
__global__ void l2norm_kernel(float* __restrict__ x, int D) {
    __shared__ float red[256];
    const size_t row = (size_t)blockIdx.x * D;
    float s = 0.f;
    for (int d = threadIdx.x; d < D; d += 256) {
        float v = x[row + d];
        s += v * v;
    }
    red[threadIdx.x] = s;
    __syncthreads();
    for (int off = 128; off > 0; off >>= 1) {
        if (threadIdx.x < off) red[threadIdx.x] += red[threadIdx.x + off];
        __syncthreads();
    }
    float scale = rsqrtf(red[0] + 1e-6f);
    for (int d = threadIdx.x; d < D; d += 256) x[row + d] *= scale;
}

// ---------------------------------------------------------------------------
// RMSNorm row -> bf16: one block per row
// ---------------------------------------------------------------------------
__global__ void rmsnorm_kernel(const float* __restrict__ x,
                               const float* __restrict__ w,
                               unsigned short* __restrict__ out, int D) {
    __shared__ float red[256];
    const size_t row = (size_t)blockIdx.x * D;
    float s = 0.f;
    for (int d = threadIdx.x; d < D; d += 256) {
        float v = x[row + d];
        s += v * v;
    }
    red[threadIdx.x] = s;
    __syncthreads();
    for (int off = 128; off > 0; off >>= 1) {
        if (threadIdx.x < off) red[threadIdx.x] += red[threadIdx.x + off];
        __syncthreads();
    }
    float scale = rsqrtf(red[0] / (float)D + 1e-5f);
    for (int d = threadIdx.x; d < D; d += 256)
        out[row + d] = f32_to_bf16(x[row + d] * scale * w[d]);
}

// ---------------------------------------------------------------------------
// LayerNorm row -> bf16: one block per row
// ---------------------------------------------------------------------------
__global__ void layernorm_kernel(const float* __restrict__ x,
                                 const float* __restrict__ g,
                                 const float* __restrict__ b,
                                 unsigned short* __restrict__ out, int D) {
    __shared__ float red0[256];
    __shared__ float red1[256];
    const size_t row = (size_t)blockIdx.x * D;
    float s = 0.f, s2 = 0.f;
    for (int d = threadIdx.x; d < D; d += 256) {
        float v = x[row + d];
        s += v;
        s2 += v * v;
    }
    red0[threadIdx.x] = s;
    red1[threadIdx.x] = s2;
    __syncthreads();
    for (int off = 128; off > 0; off >>= 1) {
        if (threadIdx.x < off) {
            red0[threadIdx.x] += red0[threadIdx.x + off];
            red1[threadIdx.x] += red1[threadIdx.x + off];
        }
        __syncthreads();
    }
    float mu  = red0[0] / (float)D;
    float var = red1[0] / (float)D - mu * mu;
    float inv = rsqrtf(var + 1e-5f);
    for (int d = threadIdx.x; d < D; d += 256)
        out[row + d] = f32_to_bf16((x[row + d] - mu) * inv * g[d] + b[d]);
}

// ---------------------------------------------------------------------------
// Delta-rule scan. State S[d,e] decomposes into independent columns over e.
// Each wave owns 2 e-columns (64 state f32/lane in VGPRs), loops t=0..S-1.
// Wave lane l covers d in [32l, 32l+32).  No LDS, no barriers.
// grid = B * (D/16) blocks of 256 threads (8 waves x 2 cols = 16 cols/block).
// ---------------------------------------------------------------------------
__global__ __launch_bounds__(256) void delta_scan_kernel(
    const float* __restrict__ q, const float* __restrict__ k,
    const float* __restrict__ v, const float* __restrict__ beta,
    float* __restrict__ o, int S, int D) {
    const int lane = threadIdx.x & 31;
    const int wave = threadIdx.x >> 5;
    const int blocksPerBatch = D / 16;
    const int b = blockIdx.x / blocksPerBatch;
    const int eBase = (blockIdx.x % blocksPerBatch) * 16 + wave * 2;
    const int d0 = lane * 32;
    const size_t base = (size_t)b * S * D;

    float Sreg[2][32];
    #pragma unroll
    for (int j = 0; j < 2; ++j)
        #pragma unroll
        for (int i = 0; i < 32; ++i) Sreg[j][i] = 0.f;

    for (int t = 0; t < S; ++t) {
        const size_t row = base + (size_t)t * D;
        // load this lane's 32-wide k slice (wave reads contiguous 4KB)
        float kv[32];
        const float4* kp = (const float4*)(k + row + d0);
        #pragma unroll
        for (int i = 0; i < 8; ++i) {
            float4 f = kp[i];
            kv[4 * i + 0] = f.x; kv[4 * i + 1] = f.y;
            kv[4 * i + 2] = f.z; kv[4 * i + 3] = f.w;
        }
        // kS = k . S[:,e]
        float p0 = 0.f, p1 = 0.f;
        #pragma unroll
        for (int i = 0; i < 32; ++i) {
            p0 += kv[i] * Sreg[0][i];
            p1 += kv[i] * Sreg[1][i];
        }
        #pragma unroll
        for (int off = 16; off > 0; off >>= 1) {
            p0 += __shfl_xor(p0, off, 32);
            p1 += __shfl_xor(p1, off, 32);
        }
        const float bt = beta[(size_t)b * S + t];
        const float dv0 = bt * (v[row + eBase]     - p0);
        const float dv1 = bt * (v[row + eBase + 1] - p1);
        // S[:,e] += k * beta*(v_e - kS_e)
        #pragma unroll
        for (int i = 0; i < 32; ++i) {
            Sreg[0][i] += kv[i] * dv0;
            Sreg[1][i] += kv[i] * dv1;
        }
        // o_e = q . S[:,e]
        float q0 = 0.f, q1 = 0.f;
        const float4* qp = (const float4*)(q + row + d0);
        #pragma unroll
        for (int i = 0; i < 8; ++i) {
            float4 f = qp[i];
            q0 += f.x * Sreg[0][4 * i + 0] + f.y * Sreg[0][4 * i + 1] +
                  f.z * Sreg[0][4 * i + 2] + f.w * Sreg[0][4 * i + 3];
            q1 += f.x * Sreg[1][4 * i + 0] + f.y * Sreg[1][4 * i + 1] +
                  f.z * Sreg[1][4 * i + 2] + f.w * Sreg[1][4 * i + 3];
        }
        #pragma unroll
        for (int off = 16; off > 0; off >>= 1) {
            q0 += __shfl_xor(q0, off, 32);
            q1 += __shfl_xor(q1, off, 32);
        }
        if (lane == 0) {
            o[row + eBase]     = q0;
            o[row + eBase + 1] = q1;
        }
    }
}

// ---------------------------------------------------------------------------
// Host-side orchestration
// ---------------------------------------------------------------------------
extern "C" void kernel_launch(void* const* d_in, const int* in_sizes, int n_in,
                              void* d_out, int out_size, void* d_ws, size_t ws_size,
                              hipStream_t stream) {
    constexpr int V = 32000, D = 1024, L = 2, B = 2, S = 1024;
    constexpr int M = B * S;   // 2048

    const int*   tokens = (const int*)d_in[0];
    const float* emb    = (const float*)d_in[1];
    const float* Wq     = (const float*)d_in[2];
    const float* Wk     = (const float*)d_in[3];
    const float* Wv     = (const float*)d_in[4];
    const float* Wb     = (const float*)d_in[5];
    const float* Wo     = (const float*)d_in[6];
    const float* rms_w  = (const float*)d_in[7];
    const float* ln_g   = (const float*)d_in[8];
    const float* ln_b   = (const float*)d_in[9];
    const float* head_w = (const float*)d_in[10];
    float* out = (float*)d_out;

    // ---- carve workspace ----
    char* ws = (char*)d_ws;
    size_t off = 0;
    auto carve = [&](size_t bytes) -> void* {
        void* p = ws + off;
        off = (off + bytes + 255) & ~(size_t)255;
        return p;
    };
    const size_t MDf = (size_t)M * D * sizeof(float);
    const size_t MDh = (size_t)M * D * sizeof(unsigned short);
    const size_t DDh = (size_t)D * D * sizeof(unsigned short);

    float*          xf   = (float*)carve(MDf);
    unsigned short* xb   = (unsigned short*)carve(MDh);
    float*          qf   = (float*)carve(MDf);
    float*          kf   = (float*)carve(MDf);
    float*          vf   = (float*)carve(MDf);
    float*          of   = (float*)carve(MDf);
    unsigned short* ob   = (unsigned short*)carve(MDh);
    unsigned short* lnb  = (unsigned short*)carve(MDh);
    float*          beta = (float*)carve((size_t)M * sizeof(float));
    unsigned short* Wqb[L]; unsigned short* Wkb[L];
    unsigned short* Wvb[L]; unsigned short* Wob[L];
    for (int l = 0; l < L; ++l) {
        Wqb[l] = (unsigned short*)carve(DDh);
        Wkb[l] = (unsigned short*)carve(DDh);
        Wvb[l] = (unsigned short*)carve(DDh);
        Wob[l] = (unsigned short*)carve(DDh);
    }
    unsigned short* headb = (unsigned short*)carve((size_t)V * D * sizeof(unsigned short));
    (void)ws_size; (void)in_sizes; (void)n_in; (void)out_size;

    // ---- 1. convert + transpose weights to bf16 [N][K] ----
    {
        dim3 blk(256);
        dim3 gDD(D / 32, D / 32);
        for (int l = 0; l < L; ++l) {
            transpose_to_bf16_kernel<<<gDD, blk, 0, stream>>>(Wq + (size_t)l * D * D, Wqb[l], D, D);
            transpose_to_bf16_kernel<<<gDD, blk, 0, stream>>>(Wk + (size_t)l * D * D, Wkb[l], D, D);
            transpose_to_bf16_kernel<<<gDD, blk, 0, stream>>>(Wv + (size_t)l * D * D, Wvb[l], D, D);
            transpose_to_bf16_kernel<<<gDD, blk, 0, stream>>>(Wo + (size_t)l * D * D, Wob[l], D, D);
        }
        dim3 gHV(V / 32, D / 32);
        transpose_to_bf16_kernel<<<gHV, blk, 0, stream>>>(head_w, headb, D, V);
    }

    // ---- 2. embedding gather ----
    embed_kernel<<<M, 256, 0, stream>>>(tokens, emb, xf, xb, D);

    // ---- 3. layers ----
    dim3 gProj(D / 128, M / 128);   // N=1024, M=2048
    for (int l = 0; l < L; ++l) {
        gemm_bf16_kernel<1, false><<<gProj, 256, 0, stream>>>(xb, Wqb[l], qf, nullptr, M, D, D);
        gemm_bf16_kernel<1, false><<<gProj, 256, 0, stream>>>(xb, Wkb[l], kf, nullptr, M, D, D);
        gemm_bf16_kernel<1, false><<<gProj, 256, 0, stream>>>(xb, Wvb[l], vf, nullptr, M, D, D);
        beta_kernel<<<M, 256, 0, stream>>>(xf, Wb + (size_t)l * D, beta, D);
        l2norm_kernel<<<M, 256, 0, stream>>>(qf, D);
        l2norm_kernel<<<M, 256, 0, stream>>>(kf, D);
        delta_scan_kernel<<<B * (D / 16), 256, 0, stream>>>(qf, kf, vf, beta, of, S, D);
        rmsnorm_kernel<<<M, 256, 0, stream>>>(of, rms_w + (size_t)l * D, ob, D);
        gemm_bf16_kernel<0, true><<<gProj, 256, 0, stream>>>(ob, Wob[l], xf, xb, M, D, D);
    }

    // ---- 4. final layernorm -> bf16 ----
    layernorm_kernel<<<M, 256, 0, stream>>>(xf, ln_g, ln_b, lnb, D);

    // ---- 5. head GEMM [2048,1024] x [1024,32000] -> d_out f32 ----
    dim3 gHead(V / 128, M / 128);   // 250 x 16
    gemm_bf16_kernel<0, false><<<gHead, 256, 0, stream>>>(lnb, headb, out, nullptr, M, V, D);
}
